// PointNet_5806795784771
// MI455X (gfx1250) — matrix operand order repaired and checked by reading, not compiled, not run on hardware
//
#include <hip/hip_runtime.h>
#include <hip/hip_bf16.h>

// ---------------------------------------------------------------------------
// PointNet forward for MI455X (gfx1250, wave32, WMMA).
// GEMMs use v_wmma_f32_16x16x32_f16 with BN scale folded into f16 weights.
// Activations staged in LDS in WMMA-B fragment-major layout so each fragment
// load is 2x ds_load_b128 per lane (no scalar u16 loads / packing moves).
// ---------------------------------------------------------------------------

typedef _Float16 v16h __attribute__((ext_vector_type(16)));
typedef _Float16 v8h  __attribute__((ext_vector_type(8)));
typedef float    v8f  __attribute__((ext_vector_type(8)));

#define BN_EPS 1e-5f
#define CB 8       // batch
#define CN 8192    // points
#define CM 256     // sampled points
#define CE 1024    // embedding
#define COUT 40
#define KNN 10

__device__ __forceinline__ v8f wmma_f16(v16h a, v16h b, v8f c) {
  // (neg_a, A, neg_b, B, c_mod, C, reuse_a, reuse_b)
  return __builtin_amdgcn_wmma_f32_16x16x32_f16(false, a, false, b, (short)0, c,
                                                false, false);
}

// A fragment (16x32, rows = output channels) from f32 weights, scaled by
// per-row BN inv factor, converted to f16.  ISA layout: lane<16 -> K 0-7
// (v0-3) & 16-23 (v4-7); lane>=16 -> K 8-15 & 24-31.  Row = lane & 15.
__device__ __forceinline__ v16h load_a_w(const float* __restrict__ W, int ldc,
                                         int row0, int k0, float scale, int lane) {
  int m = lane & 15;
  int kh = lane >> 4;
  const float* wr = W + (size_t)(row0 + m) * ldc + k0 + kh * 8;
  v16h a;
#pragma unroll
  for (int v = 0; v < 8; ++v) {
    int kb = ((v & 4) ? 16 : 0) + ((v & 3) << 1);
    a[2 * v]     = (_Float16)(wr[kb] * scale);
    a[2 * v + 1] = (_Float16)(wr[kb + 1] * scale);
  }
  return a;
}

// B fragment from LDS in fragment-major layout: tile t holds 32 lane-slots of
// 16 contiguous halves; slot(lane) = t*32 + lane.  One 32B read -> 2x b128.
__device__ __forceinline__ v16h load_b_frag(const _Float16* s, int tile, int lane) {
  return *(const v16h*)(s + ((size_t)tile * 32 + lane) * 16);
}

// ---------------------------------------------------------------------------
// Generic fused GEMM: out[o,n] = relu(inv[o]*sum_c W[o,c]*act[c,n] + beta[o])
// act: (B, C, Np) f16.  Optional f32/f16 outputs, optional max-over-n pooling.
// Block: 256 threads (8 waves): one batch, OG output channels, 128-pt chunk.
// ---------------------------------------------------------------------------
template <int CT>
__global__ void __launch_bounds__(256) gemm_bn_relu_k(
    const _Float16* __restrict__ act, const float* __restrict__ W,
    const float* __restrict__ bg, const float* __restrict__ bb,
    const float* __restrict__ bm, const float* __restrict__ bv,
    float* out_f32, _Float16* out_f16, float* maxout,
    int O, int OG, int Np) {
  constexpr int NK = CT / 32;              // K tiles
  __shared__ _Float16 smem[CT * 128];      // NK*8 tiles * 512 halves
  const int b = blockIdx.x, og = blockIdx.y, chunk = blockIdx.z;
  const int nbase = chunk * 128;
  const int tid = threadIdx.x;

  // Stage activation chunk (CT x 128) into fragment-major LDS layout.
  // Global side: coalesced 16B (b128) loads.  LDS side: 8x b16 scatter.
  for (int i = tid; i < CT * 16; i += 256) {
    int c = i >> 4;                 // channel (= K index)
    int n8 = (i & 15) * 8;          // 8-point chunk start
    v8h chunk8 = *(const v8h*)(act + ((size_t)b * CT + c) * Np + nbase + n8);
    int kk = c >> 5;                // K tile
    int khat = c & 31;              // K within tile
    int khi = (khat >> 3) & 1;      // lane half
    int e = (khat & 7) | ((khat >> 4) << 3); // element within lane slot
    int nt = n8 >> 4;               // N tile (constant for the chunk)
    int nnb = (n8 & 15) + 16 * khi; // first lane slot within tile
    _Float16* dst = smem + (((size_t)(kk * 8 + nt) * 32 + nnb) * 16 + e);
#pragma unroll
    for (int j = 0; j < 8; ++j) dst[j * 16] = chunk8[j];
  }
  __syncthreads();

  const int wave = tid >> 5, lane = tid & 31;
  const int ntow = OG >> 4;            // 4 or 8 output-channel tiles
  const int replicas = 8 / ntow;       // waves sharing one o-tile
  const int otile = wave % ntow;
  const int ntn = 8 / replicas;        // n-tiles per wave
  const int nt0 = (wave / ntow) * ntn;
  const int o0 = og * OG + otile * 16;
  const int kh = lane >> 4;

  __builtin_prefetch(W + (size_t)o0 * CT, 0, 1);

  const int lrow = o0 + (lane & 15);
  const float scale = bg[lrow] * rsqrtf(bv[lrow] + BN_EPS);
  float beta[8];
#pragma unroll
  for (int v = 0; v < 8; ++v) {
    int o = o0 + v + 8 * kh;
    float inv = bg[o] * rsqrtf(bv[o] + BN_EPS);
    beta[v] = bb[o] - bm[o] * inv;
  }

  v16h afr[NK];
#pragma unroll
  for (int kk = 0; kk < NK; ++kk)
    afr[kk] = load_a_w(W, CT, o0, kk * 32, scale, lane);

  float mxv[8];
#pragma unroll
  for (int v = 0; v < 8; ++v) mxv[v] = 0.f;

  for (int nt = nt0; nt < nt0 + ntn; ++nt) {
    v8f acc = {};
#pragma unroll
    for (int kk = 0; kk < NK; ++kk)
      acc = wmma_f16(afr[kk], load_b_frag(smem, kk * 8 + nt, lane), acc);

    float vals[8];
#pragma unroll
    for (int v = 0; v < 8; ++v) {
      vals[v] = fmaxf(acc[v] + beta[v], 0.f);
      mxv[v] = fmaxf(mxv[v], vals[v]);
    }
    int n = nbase + nt * 16 + (lane & 15);
    size_t base = ((size_t)b * O + o0 + 8 * kh) * Np + n;
    if (out_f32) {
#pragma unroll
      for (int v = 0; v < 8; ++v) out_f32[base + (size_t)v * Np] = vals[v];
    }
    if (out_f16) {
#pragma unroll
      for (int v = 0; v < 8; ++v) out_f16[base + (size_t)v * Np] = (_Float16)vals[v];
    }
  }

  if (maxout) {  // max over points; values >= 0 so int-bit atomicMax is exact
#pragma unroll
    for (int v = 0; v < 8; ++v) {
      float m = mxv[v];
#pragma unroll
      for (int off = 8; off >= 1; off >>= 1)
        m = fmaxf(m, __shfl_xor(m, off));
      if ((lane & 15) == 0) {
        int o = o0 + v + 8 * kh;
        atomicMax((int*)(maxout + (size_t)b * O + o), __float_as_int(m));
      }
    }
  }
}

// ---------------------------------------------------------------------------
// conv1 (3 -> 64) + bn1 + relu -> h0 f16.  K=3 is too small for WMMA; VALU.
// ---------------------------------------------------------------------------
__global__ void __launch_bounds__(256) conv1_bn_relu_k(
    const float* __restrict__ x, const float* __restrict__ w1,
    const float* __restrict__ bg, const float* __restrict__ bb,
    const float* __restrict__ bm, const float* __restrict__ bv,
    _Float16* __restrict__ h0) {
  __shared__ float wf[64][3];
  __shared__ float beta[64];
  int tid = threadIdx.x;
  if (tid < 64) {
    float inv = bg[tid] * rsqrtf(bv[tid] + BN_EPS);
    wf[tid][0] = w1[tid * 3 + 0] * inv;
    wf[tid][1] = w1[tid * 3 + 1] * inv;
    wf[tid][2] = w1[tid * 3 + 2] * inv;
    beta[tid] = bb[tid] - bm[tid] * inv;
  }
  __syncthreads();
  int b = blockIdx.x;
  int n = blockIdx.y * 256 + tid;
  float x0 = x[((size_t)b * 3 + 0) * CN + n];
  float x1 = x[((size_t)b * 3 + 1) * CN + n];
  float x2 = x[((size_t)b * 3 + 2) * CN + n];
#pragma unroll 4
  for (int c = 0; c < 64; ++c) {
    float v = fmaxf(x0 * wf[c][0] + x1 * wf[c][1] + x2 * wf[c][2] + beta[c], 0.f);
    h0[((size_t)b * 64 + c) * CN + n] = (_Float16)v;
  }
}

// ---------------------------------------------------------------------------
// Furthest point sampling: one 512-thread block per batch, 16 points/thread in
// registers, 256 serial steps with a shared-memory argmax (first-index ties).
// ---------------------------------------------------------------------------
#define FPS_P 16
__global__ void __launch_bounds__(512) fps_k(const float* __restrict__ xyz,
                                             int* __restrict__ idxbuf,
                                             float* __restrict__ node1) {
  __shared__ float rv[512];
  __shared__ int ri[512];
  __shared__ int s_last;
  int b = blockIdx.x, tid = threadIdx.x;
  const float* px = xyz + (size_t)b * 3 * CN;
  const float* py = px + CN;
  const float* pz = py + CN;
  float lx[FPS_P], ly[FPS_P], lz[FPS_P], dist[FPS_P];
#pragma unroll
  for (int j = 0; j < FPS_P; ++j) {
    int i = tid * FPS_P + j;
    lx[j] = px[i]; ly[j] = py[i]; lz[j] = pz[i];
    dist[j] = 1e10f;
  }
  if (tid == 0) s_last = 0;
  __syncthreads();
  for (int m = 0; m < CM; ++m) {
    int last = s_last;
    float cx = px[last], cy = py[last], cz = pz[last];
    if (tid == 0) {
      idxbuf[b * CM + m] = last;
      node1[((size_t)b * 3 + 0) * CM + m] = cx;
      node1[((size_t)b * 3 + 1) * CM + m] = cy;
      node1[((size_t)b * 3 + 2) * CM + m] = cz;
    }
    float bd = -1.f; int bi = 0;
#pragma unroll
    for (int j = 0; j < FPS_P; ++j) {
      float dx = lx[j] - cx, dy = ly[j] - cy, dz = lz[j] - cz;
      float d = dx * dx + dy * dy + dz * dz;
      dist[j] = fminf(dist[j], d);
      if (dist[j] > bd) { bd = dist[j]; bi = tid * FPS_P + j; }
    }
    rv[tid] = bd; ri[tid] = bi;
    __syncthreads();
    for (int s = 256; s > 0; s >>= 1) {
      if (tid < s) {
        float ov = rv[tid + s]; int oi = ri[tid + s];
        if (ov > rv[tid] || (ov == rv[tid] && oi < ri[tid])) { rv[tid] = ov; ri[tid] = oi; }
      }
      __syncthreads();
    }
    if (tid == 0) s_last = ri[0];
    __syncthreads();
  }
}

// ---------------------------------------------------------------------------
// kNN(k=10) aggregate + feature gather -> z0 (B,128,M) f16.
// One wave per sampled point m; per-lane register top-10, per-wave merge.
// ---------------------------------------------------------------------------
__global__ void __launch_bounds__(256) group_k(const float* __restrict__ xyz,
                                               const float* __restrict__ h,
                                               const int* __restrict__ idxbuf,
                                               _Float16* __restrict__ z0) {
  __shared__ float cd[8][320];
  __shared__ int ci[8][320];
  __shared__ int sel[8][KNN];
  int b = blockIdx.x;
  int wave = threadIdx.x >> 5, lane = threadIdx.x & 31;
  int m = blockIdx.y * 8 + wave;
  int im = idxbuf[b * CM + m];
  const float* px = xyz + (size_t)b * 3 * CN;
  const float* py = px + CN;
  const float* pz = py + CN;
  float nx = px[im], ny = py[im], nz = pz[im];
  float nn = nx * nx + ny * ny + nz * nz;
  float td[KNN]; int ti[KNN];
#pragma unroll
  for (int s = 0; s < KNN; ++s) { td[s] = 3.4e38f; ti[s] = 0x7fffffff; }
  for (int i = lane; i < CN; i += 32) {
    float qx = px[i], qy = py[i], qz = pz[i];
    float d = nn + (qx * qx + qy * qy + qz * qz) - 2.f * (qx * nx + qy * ny + qz * nz);
    if (d < td[KNN - 1]) {
      td[KNN - 1] = d; ti[KNN - 1] = i;
#pragma unroll
      for (int s = KNN - 1; s > 0; --s) {
        if (td[s] < td[s - 1]) {
          float tf = td[s]; td[s] = td[s - 1]; td[s - 1] = tf;
          int tt = ti[s]; ti[s] = ti[s - 1]; ti[s - 1] = tt;
        }
      }
    }
  }
#pragma unroll
  for (int s = 0; s < KNN; ++s) {
    cd[wave][lane * KNN + s] = td[s];
    ci[wave][lane * KNN + s] = ti[s];
  }
  __syncthreads();
  if (lane == 0) {  // merge 32 lanes x 10 candidates -> global 10 smallest
    for (int s = 0; s < KNN; ++s) {
      float bd = 3.4e38f; int bi = 0x7fffffff, bp = 0;
      for (int t = 0; t < 320; ++t) {
        float d = cd[wave][t]; int ii = ci[wave][t];
        if (d < bd || (d == bd && ii < bi)) { bd = d; bi = ii; bp = t; }
      }
      sel[wave][s] = bi;
      cd[wave][bp] = 3.4e38f;
    }
  }
  __syncthreads();
  for (int c = lane; c < 64; c += 32) {
    const float* hr = h + ((size_t)b * 64 + c) * CN;
    float nf = hr[im];
    float mx = -3.4e38f;
#pragma unroll
    for (int s = 0; s < KNN; ++s) mx = fmaxf(mx, hr[sel[wave][s]]);
    z0[((size_t)b * 128 + c) * CM + m] = (_Float16)nf;
    z0[((size_t)b * 128 + 64 + c) * CM + m] = (_Float16)mx;
  }
}

// ---------------------------------------------------------------------------
// Head: lin1 + bn6 + relu, then lin2 + bias -> logits.  Tiny (8x512, 8x40).
// ---------------------------------------------------------------------------
__global__ void __launch_bounds__(256) fc1_k(const float* __restrict__ xt1,
                                             const float* __restrict__ xt2,
                                             const float* __restrict__ lin1,
                                             const float* __restrict__ bg,
                                             const float* __restrict__ bb,
                                             const float* __restrict__ bm,
                                             const float* __restrict__ bv,
                                             float* __restrict__ g) {
  int b = blockIdx.x, tid = threadIdx.x;
  for (int j = tid; j < 512; j += 256) {
    const float* wr = lin1 + (size_t)j * (2 * CE);
    float acc = 0.f;
    for (int k = 0; k < CE; ++k) acc += wr[k] * xt1[b * CE + k];
    for (int k = 0; k < CE; ++k) acc += wr[CE + k] * xt2[b * CE + k];
    float inv = bg[j] * rsqrtf(bv[j] + BN_EPS);
    g[b * 512 + j] = fmaxf(acc * inv + (bb[j] - bm[j] * inv), 0.f);
  }
}

__global__ void __launch_bounds__(320) fc2_k(const float* __restrict__ g,
                                             const float* __restrict__ w,
                                             const float* __restrict__ bias,
                                             float* __restrict__ logits) {
  int tid = threadIdx.x;
  int b = tid / COUT, o = tid % COUT;
  const float* gr = g + b * 512;
  const float* wr = w + o * 512;
  float acc = bias[o];
  for (int k = 0; k < 512; ++k) acc += gr[k] * wr[k];
  logits[tid] = acc;
}

__global__ void zero_k(float* __restrict__ p, int n) {
  int i = blockIdx.x * 256 + threadIdx.x;
  if (i < n) p[i] = 0.f;
}

// ---------------------------------------------------------------------------
extern "C" void kernel_launch(void* const* d_in, const int* in_sizes, int n_in,
                              void* d_out, int out_size, void* d_ws, size_t ws_size,
                              hipStream_t stream) {
  const float* x     = (const float*)d_in[0];
  const float* w1    = (const float*)d_in[1];
  const float* w2    = (const float*)d_in[2];
  const float* w2m   = (const float*)d_in[3];
  const float* w3    = (const float*)d_in[4];
  const float* w4    = (const float*)d_in[5];
  const float* w5    = (const float*)d_in[6];
  const float* lin1  = (const float*)d_in[7];
  const float* lin2w = (const float*)d_in[8];
  const float* lin2b = (const float*)d_in[9];
  const float* bn1[4]  = {(const float*)d_in[10], (const float*)d_in[11],
                          (const float*)d_in[12], (const float*)d_in[13]};
  const float* bn2[4]  = {(const float*)d_in[14], (const float*)d_in[15],
                          (const float*)d_in[16], (const float*)d_in[17]};
  const float* bn2m[4] = {(const float*)d_in[18], (const float*)d_in[19],
                          (const float*)d_in[20], (const float*)d_in[21]};
  const float* bn3[4]  = {(const float*)d_in[22], (const float*)d_in[23],
                          (const float*)d_in[24], (const float*)d_in[25]};
  const float* bn4[4]  = {(const float*)d_in[26], (const float*)d_in[27],
                          (const float*)d_in[28], (const float*)d_in[29]};
  const float* bn5[4]  = {(const float*)d_in[30], (const float*)d_in[31],
                          (const float*)d_in[32], (const float*)d_in[33]};
  const float* bn6[4]  = {(const float*)d_in[34], (const float*)d_in[35],
                          (const float*)d_in[36], (const float*)d_in[37]};

  float* logits = (float*)d_out;
  float* node1  = (float*)d_out + CB * COUT;

  // Workspace carve-out (~35 MB total)
  char* p = (char*)d_ws;
  auto alloc = [&](size_t bytes) -> void* {
    void* r = (void*)p;
    p += (bytes + 255) & ~(size_t)255;
    return r;
  };
  _Float16* h0   = (_Float16*)alloc((size_t)CB * 64 * CN * 2);
  float*    hf32 = (float*)   alloc((size_t)CB * 64 * CN * 4);
  _Float16* hf16 = (_Float16*)alloc((size_t)CB * 64 * CN * 2);
  float*    xt1  = (float*)   alloc((size_t)CB * CE * 4);
  float*    xt2  = (float*)   alloc((size_t)CB * CE * 4);  // adjacent to xt1
  int*      idxb = (int*)     alloc((size_t)CB * CM * 4);
  _Float16* z0   = (_Float16*)alloc((size_t)CB * 128 * CM * 2);
  _Float16* z1   = (_Float16*)alloc((size_t)CB * 128 * CM * 2);
  _Float16* z2   = (_Float16*)alloc((size_t)CB * 128 * CM * 2);
  float*    g    = (float*)   alloc((size_t)CB * 512 * 4);

  // 0) zero the max-pool accumulators (xt1, xt2 contiguous: 16384 floats)
  zero_k<<<dim3(64), dim3(256), 0, stream>>>(xt1, 2 * CB * CE);

  // 1) conv1 + bn1 + relu -> h0 f16
  conv1_bn_relu_k<<<dim3(CB, CN / 256), dim3(256), 0, stream>>>(
      x, w1, bn1[0], bn1[1], bn1[2], bn1[3], h0);

  // 2) conv2 + bn2 + relu -> h (f32 + f16)   [WMMA, C=64, O=64]
  gemm_bn_relu_k<64><<<dim3(CB, 1, CN / 128), dim3(256), 0, stream>>>(
      h0, w2, bn2[0], bn2[1], bn2[2], bn2[3], hf32, hf16, nullptr, 64, 64, CN);

  // 3) w2m + bn2m + relu, fused max over N -> xt1   [WMMA, C=64, O=1024]
  gemm_bn_relu_k<64><<<dim3(CB, CE / 128, CN / 128), dim3(256), 0, stream>>>(
      hf16, w2m, bn2m[0], bn2m[1], bn2m[2], bn2m[3], nullptr, nullptr, xt1,
      CE, 128, CN);

  // 4) furthest point sampling -> idx, node1 (second output)
  fps_k<<<dim3(CB), dim3(512), 0, stream>>>(x, idxb, node1);

  // 5) gather + kNN-max aggregate -> z0 (B,128,M) f16
  group_k<<<dim3(CB, CM / 8), dim3(256), 0, stream>>>(x, hf32, idxb, z0);

  // 6) conv3 -> z1, conv4 -> z2   [WMMA, C=128, O=128]
  gemm_bn_relu_k<128><<<dim3(CB, 1, CM / 128), dim3(256), 0, stream>>>(
      z0, w3, bn3[0], bn3[1], bn3[2], bn3[3], nullptr, z1, nullptr, 128, 128, CM);
  gemm_bn_relu_k<128><<<dim3(CB, 1, CM / 128), dim3(256), 0, stream>>>(
      z1, w4, bn4[0], bn4[1], bn4[2], bn4[3], nullptr, z2, nullptr, 128, 128, CM);

  // 7) w5 + bn5 + relu, fused max over M -> xt2   [WMMA, C=128, O=1024]
  gemm_bn_relu_k<128><<<dim3(CB, CE / 128, CM / 128), dim3(256), 0, stream>>>(
      z2, w5, bn5[0], bn5[1], bn5[2], bn5[3], nullptr, nullptr, xt2, CE, 128, CM);

  // 8) head
  fc1_k<<<dim3(CB), dim3(256), 0, stream>>>(xt1, xt2, lin1, bn6[0], bn6[1],
                                            bn6[2], bn6[3], g);
  fc2_k<<<dim3(1), dim3(320), 0, stream>>>(g, lin2w, lin2b, logits);
}
